// MTA_StableWeight_58634893525401
// MI455X (gfx1250) — matrix operand order repaired and checked by reading, not compiled, not run on hardware
//
#include <hip/hip_runtime.h>

// ---------------------------------------------------------------------------
// Fused MoE gated-GELU FFN (expert weights all == 1 per the reference's
// softmax-of-scalar bug):   Y = sum_e gelu_tanh(X@W0_e) * (X@W1_e) @ Wo_e
//
// Compute-bound (AI ~1800 FLOP/B). Pipeline:
//   1) transpose+convert weights to bf16 in workspace (W^T: K contiguous per
//      output column) so WMMA B-fragments are 2x16B contiguous loads per lane
//   2) fused main kernel: 512 thr / 16 waves, M-tile 64, FF split x4 across
//      blockIdx.y, Y partials (fp32) to workspace
//   3) deterministic reduction of the 4 FF-slice partials into d_out
// ---------------------------------------------------------------------------

typedef __attribute__((ext_vector_type(16))) __bf16 v16bf;
typedef __attribute__((ext_vector_type(8)))  __bf16 v8bf;
typedef __attribute__((ext_vector_type(8)))  float  v8f;

#define DMODEL 1024
#define DFF    4096
#define NEXP   5
#define MROWS  4096          // B*S
#define MTILE  64            // rows per workgroup
#define FSLICES 4            // FF split across blockIdx.y
#define FSLICE (DFF / FSLICES)   // 1024 ff-cols per slice
#define FCHUNK 128           // ff cols per fused chunk (8 waves x 16)
#define XS_LD  1032          // 1024 + 8 bf16 pad (rows stay 16B aligned)
#define HS_LD  136           // 128 + 8 bf16 pad

// workspace layout (bytes)
#define WT_ELEMS   ((size_t)NEXP * DMODEL * DFF)          // 20,971,520
#define W0T_OFF    ((size_t)0)
#define W1T_OFF    (WT_ELEMS * 2)                          // 41,943,040
#define WOT_OFF    (WT_ELEMS * 4)                          // 83,886,080
#define PART_OFF   (WT_ELEMS * 6)                          // 125,829,120
#define SLICE_F32  ((size_t)MROWS * DMODEL)                // 4,194,304 floats

__device__ __forceinline__ float gelu_tanh(float x) {
    // tanh approx GELU (jax approximate=True); tanh(y) = 1 - 2/(exp(2y)+1)
    const float c0 = 0.7978845608028654f, c1 = 0.044715f;
    float y = c0 * (x + c1 * x * x * x);
    float t = 1.0f - 2.0f / (__expf(2.0f * y) + 1.0f);
    return 0.5f * x * (1.0f + t);
}

// A fragment (16x32 bf16) from LDS row-major [m][k], stride ld (bf16 elems).
// lanes 0-15 hold K k0+0..7 & k0+16..23; lanes 16-31 hold K k0+8..15 & 24..31.
__device__ __forceinline__ v16bf load_a_lds(const __bf16* s, int row0, int k0,
                                            int ld, int lane) {
    int half = lane >> 4, r = lane & 15;
    const __bf16* p = s + (size_t)(row0 + r) * ld + k0 + half * 8;
    v8bf lo = *reinterpret_cast<const v8bf*>(p);
    v8bf hi = *reinterpret_cast<const v8bf*>(p + 16);
    v16bf a;
#pragma unroll
    for (int i = 0; i < 8; ++i) { a[i] = lo[i]; a[8 + i] = hi[i]; }
    return a;
}

// B fragment (32x16 bf16) from pre-transposed bf16 weights W^T[n][k]:
// p must point at (column n for this lane, k0 + half*8) -> K +0..7 & +16..23.
__device__ __forceinline__ v16bf load_b_t(const __bf16* p) {
    v8bf lo = *reinterpret_cast<const v8bf*>(p);
    v8bf hi = *reinterpret_cast<const v8bf*>(p + 16);
    v16bf b;
#pragma unroll
    for (int i = 0; i < 8; ++i) { b[i] = lo[i]; b[8 + i] = hi[i]; }
    return b;
}

__device__ __forceinline__ v8f wmma_bf16(v16bf a, v16bf b, v8f c) {
    return __builtin_amdgcn_wmma_f32_16x16x32_bf16(false, a, false, b,
                                                   (short)0, c, false, false);
}

// ---- Kernel 1: transpose fp32 [R][C] -> bf16 [C][R], batched over z ----
__global__ __launch_bounds__(256)
void transpose_bf16_kernel(const float* __restrict__ src,
                           __bf16* __restrict__ dst, int R, int C) {
    __shared__ float t[32][33];
    const float* s = src + (size_t)blockIdx.z * R * C;
    __bf16*      d = dst + (size_t)blockIdx.z * R * C;
    int c0 = blockIdx.x * 32, r0 = blockIdx.y * 32;
#pragma unroll
    for (int i = threadIdx.y; i < 32; i += 8)
        t[i][threadIdx.x] = s[(size_t)(r0 + i) * C + c0 + threadIdx.x];
    __syncthreads();
#pragma unroll
    for (int i = threadIdx.y; i < 32; i += 8)
        d[(size_t)(c0 + i) * R + r0 + threadIdx.x] = (__bf16)t[threadIdx.x][i];
}

// ---- Kernel 2: fused MoE FFN over one (M-block, FF-slice) ----
__global__ __launch_bounds__(512, 1)
void moe_main_kernel(const float* __restrict__ X,
                     const __bf16* __restrict__ w0t,   // [E][f][k]
                     const __bf16* __restrict__ w1t,   // [E][f][k]
                     const __bf16* __restrict__ wot,   // [E][d][f]
                     float* __restrict__ part) {       // [FSLICES][M][D]
    __shared__ __align__(16) __bf16 Xs[MTILE * XS_LD];  // 64x1024 bf16 X tile
    __shared__ __align__(16) __bf16 Hs[MTILE * HS_LD];  // 64x128 bf16 H chunk

    const int tid  = threadIdx.x;
    const int wave = tid >> 5;
    const int lane = tid & 31;
    const int half = lane >> 4;
    const int ln   = lane & 15;
    const int mgrp = wave >> 3;   // 0/1: rows mgrp*32 .. +32 (2 WMMA M-tiles)
    const int sub  = wave & 7;    // f-tile in GEMM1 / d-group in GEMM2
    const int r0   = blockIdx.x * MTILE;
    const int fs   = blockIdx.y;  // FF slice

    // stage X tile as bf16 (64 rows x 256 float4)
    for (int i = tid; i < MTILE * (DMODEL / 4); i += 512) {
        int r = i >> 8, c4 = i & 255;
        float4 x = reinterpret_cast<const float4*>(
            X + (size_t)(r0 + r) * DMODEL)[c4];
        __bf16* d = Xs + (size_t)r * XS_LD + c4 * 4;
        d[0] = (__bf16)x.x; d[1] = (__bf16)x.y;
        d[2] = (__bf16)x.z; d[3] = (__bf16)x.w;
    }
    __syncthreads();

    const v8f vzero = {0.f, 0.f, 0.f, 0.f, 0.f, 0.f, 0.f, 0.f};
    v8f yacc[2][8];               // [mtile][dtile]: rows mgrp*32+mt*16, cols sub*128+t*16
#pragma unroll
    for (int mt = 0; mt < 2; ++mt)
#pragma unroll
        for (int t = 0; t < 8; ++t) yacc[mt][t] = vzero;

    for (int e = 0; e < NEXP; ++e) {
        const __bf16* W0 = w0t + (size_t)e * DFF * DMODEL;
        const __bf16* W1 = w1t + (size_t)e * DFF * DMODEL;
        const __bf16* Wo = wot + (size_t)e * DMODEL * DFF;

        for (int f0 = fs * FSLICE; f0 < (fs + 1) * FSLICE; f0 += FCHUNK) {
            // ---- GEMM1: G/L tiles for f-cols f0+sub*16, rows mgrp*32..+32 ----
            v8f g[2] = {vzero, vzero}, l[2] = {vzero, vzero};
            {
                const int fcol = f0 + sub * 16 + ln;          // this lane's B col
                const __bf16* b0p = W0 + (size_t)fcol * DMODEL + half * 8;
                const __bf16* b1p = W1 + (size_t)fcol * DMODEL + half * 8;
                for (int k0 = 0; k0 < DMODEL; k0 += 32) {
                    v16bf b0 = load_b_t(b0p + k0);
                    v16bf b1 = load_b_t(b1p + k0);
#pragma unroll
                    for (int mt = 0; mt < 2; ++mt) {
                        v16bf a = load_a_lds(Xs, mgrp * 32 + mt * 16, k0,
                                             XS_LD, lane);
                        g[mt] = wmma_bf16(a, b0, g[mt]);
                        l[mt] = wmma_bf16(a, b1, l[mt]);
                    }
                }
            }
            // ---- H = gelu(G)*L -> LDS; C layout: (m = v+8*half, n = ln) ----
#pragma unroll
            for (int mt = 0; mt < 2; ++mt) {
                __bf16* hp = Hs + (size_t)(mgrp * 32 + mt * 16 + 8 * half) * HS_LD
                                + sub * 16 + ln;
#pragma unroll
                for (int v = 0; v < 8; ++v)
                    hp[(size_t)v * HS_LD] = (__bf16)(gelu_tanh(g[mt][v]) * l[mt][v]);
            }
            __syncthreads();

            // ---- GEMM2: Y[mgrp rows][sub*128..+128] += H(64x128) @ Wo^T ----
            for (int k0 = 0; k0 < FCHUNK; k0 += 32) {
                v16bf a0 = load_a_lds(Hs, mgrp * 32,      k0, HS_LD, lane);
                v16bf a1 = load_a_lds(Hs, mgrp * 32 + 16, k0, HS_LD, lane);
#pragma unroll
                for (int t = 0; t < 8; ++t) {
                    int dcol = sub * 128 + t * 16 + ln;
                    v16bf b = load_b_t(Wo + (size_t)dcol * DFF
                                          + (f0 + k0) + half * 8);
                    yacc[0][t] = wmma_bf16(a0, b, yacc[0][t]);
                    yacc[1][t] = wmma_bf16(a1, b, yacc[1][t]);
                }
            }
            __syncthreads();   // before next chunk overwrites Hs
        }
    }

    // ---- store fp32 partials for this FF slice ----
    float* P = part + (size_t)fs * SLICE_F32;
#pragma unroll
    for (int mt = 0; mt < 2; ++mt) {
        int row0 = r0 + mgrp * 32 + mt * 16 + 8 * half;
#pragma unroll
        for (int t = 0; t < 8; ++t) {
            int col = sub * 128 + t * 16 + ln;
#pragma unroll
            for (int v = 0; v < 8; ++v)
                P[(size_t)(row0 + v) * DMODEL + col] = yacc[mt][t][v];
        }
    }
}

// ---- Kernel 3: Y = sum of 4 FF-slice partials (deterministic) ----
__global__ __launch_bounds__(256)
void reduce_slices_kernel(const float* __restrict__ part,
                          float* __restrict__ y) {
    size_t i = (size_t)blockIdx.x * blockDim.x + threadIdx.x;  // float4 index
    const size_t s4 = SLICE_F32 / 4;
    const float4* p4 = reinterpret_cast<const float4*>(part);
    float4 a = p4[i], b = p4[i + s4], c = p4[i + 2 * s4], d = p4[i + 3 * s4];
    float4 r;
    r.x = (a.x + b.x) + (c.x + d.x);
    r.y = (a.y + b.y) + (c.y + d.y);
    r.z = (a.z + b.z) + (c.z + d.z);
    r.w = (a.w + b.w) + (c.w + d.w);
    reinterpret_cast<float4*>(y)[i] = r;
}

extern "C" void kernel_launch(void* const* d_in, const int* in_sizes, int n_in,
                              void* d_out, int out_size, void* d_ws, size_t ws_size,
                              hipStream_t stream) {
    (void)in_sizes; (void)n_in; (void)out_size; (void)ws_size;
    const float* X   = (const float*)d_in[0];
    // d_in[1] = type_label: dead in the reference (softmax of scalar == 1.0)
    const float* wi0 = (const float*)d_in[2];
    const float* wi1 = (const float*)d_in[3];
    const float* wo  = (const float*)d_in[4];
    float* Y = (float*)d_out;

    char* ws = (char*)d_ws;
    __bf16* w0t = (__bf16*)(ws + W0T_OFF);
    __bf16* w1t = (__bf16*)(ws + W1T_OFF);
    __bf16* wot = (__bf16*)(ws + WOT_OFF);
    float*  prt = (float*)(ws + PART_OFF);

    dim3 tb(32, 8);
    // wi0/wi1: [k=1024][f=4096] -> [f][k]
    transpose_bf16_kernel<<<dim3(DFF / 32, DMODEL / 32, NEXP), tb, 0, stream>>>(
        wi0, w0t, DMODEL, DFF);
    transpose_bf16_kernel<<<dim3(DFF / 32, DMODEL / 32, NEXP), tb, 0, stream>>>(
        wi1, w1t, DMODEL, DFF);
    // wo: [f=4096][d=1024] -> [d][f]
    transpose_bf16_kernel<<<dim3(DMODEL / 32, DFF / 32, NEXP), tb, 0, stream>>>(
        wo, wot, DFF, DMODEL);

    moe_main_kernel<<<dim3(MROWS / MTILE, FSLICES), 512, 0, stream>>>(
        X, w0t, w1t, wot, prt);

    reduce_slices_kernel<<<(SLICE_F32 / 4) / 256, 256, 0, stream>>>(prt, Y);
}
